// APDL_RNN_18897856103016
// MI455X (gfx1250) — matrix operand order repaired and checked by reading, not compiled, not run on hardware
//
#include <hip/hip_runtime.h>

#define HID  64
#define BSZ  32
#define TL   16384
#define SDIM 128   // 2*HIDDEN

typedef _Float16 v16h __attribute__((ext_vector_type(16)));
typedef float    v8f  __attribute__((ext_vector_type(8)));

union HFrag { v16h v; unsigned int u[8]; _Float16 h[16]; };

// Fast activations: v_exp_f32 + v_rcp_f32 (no IEEE div expansion).
__device__ __forceinline__ float frcp(float z)  { return __builtin_amdgcn_rcpf(z); }
__device__ __forceinline__ float fsig(float z)  { return frcp(1.0f + __expf(-z)); }
__device__ __forceinline__ float ftanh(float z) { return 1.0f - 2.0f * frcp(1.0f + __expf(2.0f * z)); }

// One persistent workgroup: 8 waves x 32 lanes.
// wave = (mtile, cgrp): mtile picks rows 16*mtile..+15 of the 32-row batch,
// cgrp picks hidden columns 16*cgrp..+15 (same columns for all 4 gates).
__global__ __launch_bounds__(256, 1)
void apdl_lstm_wmma(const float* __restrict__ x,
                    const float* __restrict__ W_ih,
                    const float* __restrict__ W_hh,
                    const float* __restrict__ b_ih,
                    const float* __restrict__ b_hh,
                    float* __restrict__ out)
{
    // h stored directly in WMMA A-fragment order:
    // half index = (mtile*32 + lane)*32 + frag*16 + j   (frag 0: K0..31, frag 1: K32..63)
    __shared__ _Float16 hfA[2 * 32 * 32];   // 4 KB
    __shared__ float    xbuf[BSZ];          // x[:, t]

    const int tid   = threadIdx.x;
    const int lane  = tid & 31;
    const int wave  = tid >> 5;
    const int g     = lane >> 4;          // lane half-group (ISA fragment layouts)
    const int r     = lane & 15;
    const int mtile = wave & 1;
    const int cgrp  = wave >> 1;          // 0..3
    const int mbase = mtile << 4;
    const int k     = (cgrp << 4) + r;    // hidden column 0..63 (within each gate)

    float* yo0 = out;                                    // y
    float* yo1 = out + (size_t)1 * BSZ * TL * HID;       // y (tuple copy)
    float* co  = out + (size_t)2 * BSZ * TL * HID;       // c states
    float* apo = out + (size_t)3 * BSZ * TL * HID;       // ap_final (B x 128)

    // Per-lane gate constants (fixed column k), loaded once.
    const float bI = b_ih[        k] + b_hh[        k];
    const float bF = b_ih[  HID + k] + b_hh[  HID + k];
    const float bG = b_ih[2*HID + k] + b_hh[2*HID + k];
    const float bO = b_ih[3*HID + k] + b_hh[3*HID + k];
    const float wI = W_ih[k], wF = W_ih[HID + k], wG = W_ih[2*HID + k], wO = W_ih[3*HID + k];

    // B fragments resident in VGPRs for the whole kernel.
    // B[K][n] = W_hh[n][K]; lane holds column n = G*64 + k,
    // halves j=0..15 -> K = kb + 16*g + j (K,K+1 packed per VGPR).
    HFrag Bf[4][2];
#pragma unroll
    for (int G = 0; G < 4; ++G) {
        const int n = G * HID + k;
#pragma unroll
        for (int kh = 0; kh < 2; ++kh) {
            const int kb = kh * 32 + (g << 4);
#pragma unroll
            for (int j = 0; j < 16; ++j)
                Bf[G][kh].h[j] = (_Float16)W_hh[n * HID + kb + j];
        }
    }

    // Writer-side target slot for h(b, k): k fixed per lane -> frag/group/j fixed.
    const int wf  = k >> 5;                                  // which A fragment
    const int Kof = k & 31;
    const int wgp = Kof >> 4;                                // lane half-group at reader
    const int wj  = Kof & 15;                                // half slot within fragment
    // half index = mtile*1024 + (b&15)*32 + wgp*512 + wf*16 + wj ; (b&15) = 8g+v
    const int wbase = mtile * 1024 + (wgp << 9) + (wf << 4) + wj + ((g << 3) << 5);

    // Recurrent state in registers, ownership aligned with the WMMA C layout:
    // q = 0..7  -> (row b = mbase + 8g + v, col k)      : h part
    // q = 8..15 -> (row b, col 64 + k)                  : c part
    float s_m1[16], s_m2[16], ap[16];
#pragma unroll
    for (int q = 0; q < 16; ++q) { s_m1[q] = 0.f; s_m2[q] = 0.f; ap[q] = 0.f; }

    const unsigned int* hfU = (const unsigned int*)hfA;
    const int dbase = (mtile * 32 + lane) * 16;   // reader: dword base of this lane's fragments
    const float coeff = 0.8375f;  // (1-a)/(1+a), a = 48000/44100 - 1 = 13/147

    for (int t = 0; t < TL; ++t) {
        // ---- allpass state update (pure registers) + export h to LDS (A-fragment order)
        float apn[16];
#pragma unroll
        for (int q = 0; q < 16; ++q) {
            apn[q] = coeff * (s_m1[q] - ap[q]) + s_m2[q];
            ap[q]  = apn[q];
        }
#pragma unroll
        for (int v = 0; v < 8; ++v)
            hfA[wbase + v * 32] = (_Float16)apn[v];
        if (tid < BSZ) xbuf[tid] = x[tid * TL + t];
        __syncthreads();

        // ---- A fragments: contiguous 16 dwords per lane -> b128 LDS loads
        HFrag A0, A1;
#pragma unroll
        for (int p = 0; p < 8; ++p) {
            A0.u[p] = hfU[dbase + p];
            A1.u[p] = hfU[dbase + 8 + p];
        }

        // ---- gates = h @ W_hh^T : one 16x16 tile per gate, K=64 as 2x K32 chain
        v8f acc[4];
#pragma unroll
        for (int G = 0; G < 4; ++G) {
            v8f z = {0.f, 0.f, 0.f, 0.f, 0.f, 0.f, 0.f, 0.f};
            z = __builtin_amdgcn_wmma_f32_16x16x32_f16(false, A0.v, false, Bf[G][0].v,
                                                       (short)0, z, false, false);
            z = __builtin_amdgcn_wmma_f32_16x16x32_f16(false, A1.v, false, Bf[G][1].v,
                                                       (short)0, z, false, false);
            acc[G] = z;
        }

        // ---- LSTM pointwise directly on accumulator fragments (registers only)
        float s_new[16];
#pragma unroll
        for (int v = 0; v < 8; ++v) {
            const int   b     = mbase + (g << 3) + v;
            const float xv    = xbuf[b];
            const float iv    = fsig (acc[0][v] + bI + xv * wI);
            const float fv    = fsig (acc[1][v] + bF + xv * wF);
            const float gv    = ftanh(acc[2][v] + bG + xv * wG);
            const float ov    = fsig (acc[3][v] + bO + xv * wO);
            const float c_old = apn[8 + v];                 // c from ap_new, same lane
            const float c_new = fv * c_old + iv * gv;
            const float h_new = ov * ftanh(c_new);
            s_new[v]     = h_new;
            s_new[8 + v] = c_new;
            const size_t o = ((size_t)b * TL + t) * HID + k;
            __builtin_nontemporal_store(h_new, yo0 + o);
            __builtin_nontemporal_store(h_new, yo1 + o);
            __builtin_nontemporal_store(c_new, co  + o);
        }

        // rotate recurrence: s_m2 <- s_m1 <- s_new
#pragma unroll
        for (int q = 0; q < 16; ++q) { s_m2[q] = s_m1[q]; s_m1[q] = s_new[q]; }
        __syncthreads();   // protect hfA/xbuf for next step's rewrite
    }

    // ---- ap_final (B x 128), straight from registers
#pragma unroll
    for (int v = 0; v < 8; ++v) {
        const int b = mbase + (g << 3) + v;
        apo[b * SDIM +       k] = ap[v];
        apo[b * SDIM + HID + k] = ap[8 + v];
    }
}

extern "C" void kernel_launch(void* const* d_in, const int* in_sizes, int n_in,
                              void* d_out, int out_size, void* d_ws, size_t ws_size,
                              hipStream_t stream) {
    (void)in_sizes; (void)n_in; (void)out_size; (void)d_ws; (void)ws_size;
    const float* x    = (const float*)d_in[0];
    const float* W_ih = (const float*)d_in[1];
    const float* W_hh = (const float*)d_in[2];
    const float* b_ih = (const float*)d_in[3];
    const float* b_hh = (const float*)d_in[4];
    float* out = (float*)d_out;
    // Single persistent workgroup: the scan is strictly sequential in time;
    // all parallelism (B=32 rows x 256 gate columns) fits in 8 wave32 waves.
    apdl_lstm_wmma<<<dim3(1), dim3(256), 0, stream>>>(x, W_ih, W_hh, b_ih, b_hh, out);
}
